// JustGAT_51170240364904
// MI455X (gfx1250) — matrix operand order repaired and checked by reading, not compiled, not run on hardware
//
#include <hip/hip_runtime.h>

// ---------------------------------------------------------------------------
// GATv2 (2 layers) for MI455X / gfx1250, wave32 + WMMA f16->f32.
// Weights are pre-swizzled once per layer into WMMA B-fragment layout so the
// hot GEMM loop is pure b128 loads + v_wmma.
// ---------------------------------------------------------------------------

typedef __attribute__((ext_vector_type(16))) _Float16 v16h;
typedef __attribute__((ext_vector_type(8)))  float    v8f;

#define N_NODES   100000
#define E_EDGES   640000
#define C_DIM     128
#define E_TOT     (E_EDGES + N_NODES)   // 740000, = 92500 * 8
#define NEG_SLOPE 0.2f
#define WPACK_DW  8192                  // dwords per packed weight matrix

union AFrag { v16h h; unsigned u[8]; uint4 q[2]; };
union BFrag { v16h h; unsigned u[8]; uint4 q[2]; };
union CFrag { v8f  v; float    f[8]; };

__device__ inline unsigned pack2h(float a, float b) {
    union { _Float16 h[2]; unsigned u; } p;
    p.h[0] = (_Float16)a;
    p.h[1] = (_Float16)b;
    return p.u;
}

// float atomic max via sign-split native int atomics (no CAS loop).
__device__ inline void atomicMaxFloat(float* addr, float val) {
    if (val >= 0.0f) {
        atomicMax((int*)addr, __float_as_int(val));
    } else {
        atomicMin((unsigned int*)addr, __float_as_uint(val));
    }
}

// ---------------------------------------------------------------------------
// Kernel 0: pre-swizzle the three weight matrices into per-lane WMMA
// B-fragment layout (f16).  Dword index:  ((ct*4 + kc)*32 + lane)*8 + v
//   n = ct*16 + (lane&15) ;  k = kc*32 + 16*(lane>>4) + 2*v
//   dword = { W[n][k], W[n][k+1] } as packed f16  (B[k][n] = W[n][k])
// ---------------------------------------------------------------------------
__global__ __launch_bounds__(256)
void gat_pack_weights(const float* __restrict__ Wl, const float* __restrict__ Wr,
                      const float* __restrict__ Wres,
                      unsigned* __restrict__ pl, unsigned* __restrict__ pr,
                      unsigned* __restrict__ ps)
{
    const int i    = blockIdx.x * 256 + threadIdx.x;  // 32 blocks -> 8192
    const int v    = i & 7;
    const int lane = (i >> 3) & 31;
    const int kc   = (i >> 8) & 3;
    const int ct   = i >> 10;
    const int n = ct * 16 + (lane & 15);
    const int k = kc * 32 + 16 * (lane >> 4) + 2 * v;
    const int src = n * C_DIM + k;
    pl[i] = pack2h(Wl  [src], Wl  [src + 1]);
    pr[i] = pack2h(Wr  [src], Wr  [src + 1]);
    ps[i] = pack2h(Wres[src], Wres[src + 1]);
}

// ---------------------------------------------------------------------------
// Kernel 1: node transform.  xl = x@Wl^T + bl ; xr = x@Wr^T + br ;
//           res = x@Wres^T + bias.   One block = 16 node rows, 8 waves,
//           wave w owns output columns [16w, 16w+16).  K = 128 in 4 chunks.
// ---------------------------------------------------------------------------
__global__ __launch_bounds__(256)
void gat_transform(const float* __restrict__ x,
                   const unsigned* __restrict__ pl, const unsigned* __restrict__ pr,
                   const unsigned* __restrict__ ps,
                   const float* __restrict__ bl,  const float* __restrict__ br,
                   const float* __restrict__ bias,
                   float* __restrict__ xl, float* __restrict__ xr,
                   float* __restrict__ res)
{
    __shared__ _Float16 sx[16 * C_DIM];     // 4 KB: x tile as f16

    const int row0 = blockIdx.x * 16;       // 6250 * 16 == 100000 exactly
    const int tid  = threadIdx.x;

    // stage 16x128 f32 -> f16 into LDS: 8 contiguous elems / thread
    {
        const float* src = &x[row0 * C_DIM + tid * 8];
        const float4 f0 = *(const float4*)(src);
        const float4 f1 = *(const float4*)(src + 4);
        union { unsigned u[4]; uint4 q; } w;
        w.u[0] = pack2h(f0.x, f0.y);
        w.u[1] = pack2h(f0.z, f0.w);
        w.u[2] = pack2h(f1.x, f1.y);
        w.u[3] = pack2h(f1.z, f1.w);
        *(uint4*)(&sx[tid * 8]) = w.q;
    }
    __syncthreads();

    const int lane = tid & 31;
    const int wv   = tid >> 5;              // 0..7 : output col tile = 16*wv
    const int m    = lane & 15;             // A row / B,C,D col n
    const int half = lane >> 4;

    CFrag cl, cr, cs;
    #pragma unroll
    for (int v = 0; v < 8; ++v) { cl.f[v] = 0.f; cr.f[v] = 0.f; cs.f[v] = 0.f; }

    #pragma unroll
    for (int kc = 0; kc < 4; ++kc) {
        // A fragment (16x32 f16): k = kc*32 + 16*half + 2*v
        //   -> v=0..7 are 32 contiguous bytes in LDS.
        AFrag a;
        const _Float16* ap = &sx[m * C_DIM + kc * 32 + 16 * half];
        a.q[0] = *(const uint4*)(ap);
        a.q[1] = *(const uint4*)(ap + 8);

        // B fragments: pre-swizzled, 8 contiguous dwords per lane.
        const int boff = ((wv * 4 + kc) * 32 + lane) * 8;
        BFrag bL, bR, bS;
        bL.q[0] = *(const uint4*)&pl[boff];  bL.q[1] = *(const uint4*)&pl[boff + 4];
        bR.q[0] = *(const uint4*)&pr[boff];  bR.q[1] = *(const uint4*)&pr[boff + 4];
        bS.q[0] = *(const uint4*)&ps[boff];  bS.q[1] = *(const uint4*)&ps[boff + 4];

        cl.v = __builtin_amdgcn_wmma_f32_16x16x32_f16(false, a.h, false, bL.h,
                                                      (short)0, cl.v, false, false);
        cr.v = __builtin_amdgcn_wmma_f32_16x16x32_f16(false, a.h, false, bR.h,
                                                      (short)0, cr.v, false, false);
        cs.v = __builtin_amdgcn_wmma_f32_16x16x32_f16(false, a.h, false, bS.h,
                                                      (short)0, cs.v, false, false);
    }

    // epilogue: C/D layout -> lane gives n, VGPR v gives M = v + 8*half
    const int n0 = wv * 16;
    const float addl  = bl  [n0 + m];
    const float addr_ = br  [n0 + m];
    const float adds  = bias[n0 + m];
    #pragma unroll
    for (int v = 0; v < 8; ++v) {
        const int row = row0 + v + 8 * half;
        xl [row * C_DIM + n0 + m] = cl.f[v] + addl;
        xr [row * C_DIM + n0 + m] = cr.f[v] + addr_;
        res[row * C_DIM + n0 + m] = cs.f[v] + adds;
    }
}

// ---------------------------------------------------------------------------
// Kernel 2: init segment state.  12.8M threads: zero acc; first N set
//           emax=-inf, denom=0.
// ---------------------------------------------------------------------------
__global__ __launch_bounds__(256)
void gat_init(float* __restrict__ emax, float* __restrict__ denom,
              float* __restrict__ acc)
{
    const int i = blockIdx.x * 256 + threadIdx.x;   // 50000*256 == N*128 exactly
    acc[i] = 0.0f;
    if (i < N_NODES) {
        emax[i]  = __uint_as_float(0xFF800000u);    // -inf
        denom[i] = 0.0f;
    }
}

// ---------------------------------------------------------------------------
// Kernel 3: per-edge attention logit + segment max.  One wave32 per edge;
//           lane covers 4 channels (float4).
// ---------------------------------------------------------------------------
__global__ __launch_bounds__(256)
void gat_edge_logits(const int* __restrict__ ei,
                     const float* __restrict__ xl, const float* __restrict__ xr,
                     const float* __restrict__ att,
                     float* __restrict__ e_out, float* __restrict__ emax)
{
    const int edge = (blockIdx.x * 256 + threadIdx.x) >> 5;  // 92500*8 == E_TOT
    const int lane = threadIdx.x & 31;

    int s, d;
    if (edge < E_EDGES) { s = ei[edge]; d = ei[E_EDGES + edge]; }
    else                { s = edge - E_EDGES; d = s; }

    const float4 xs = *(const float4*)&xl [s * C_DIM + lane * 4];
    const float4 xd = *(const float4*)&xr [d * C_DIM + lane * 4];
    const float4 av = *(const float4*)&att[lane * 4];

    float p = 0.0f, t;
    t = xs.x + xd.x; p += av.x * (t > 0.0f ? t : NEG_SLOPE * t);
    t = xs.y + xd.y; p += av.y * (t > 0.0f ? t : NEG_SLOPE * t);
    t = xs.z + xd.z; p += av.z * (t > 0.0f ? t : NEG_SLOPE * t);
    t = xs.w + xd.w; p += av.w * (t > 0.0f ? t : NEG_SLOPE * t);

    #pragma unroll
    for (int off = 16; off > 0; off >>= 1)
        p += __shfl_xor(p, off, 32);

    if (lane == 0) {
        e_out[edge] = p;
        atomicMaxFloat(&emax[d], p);
    }
}

// ---------------------------------------------------------------------------
// Kernel 4: unnormalized segment-softmax accumulate.
//           acc[dst] += exp(e - emax[dst]) * xl[src];  denom[dst] += exp(...)
// ---------------------------------------------------------------------------
__global__ __launch_bounds__(256)
void gat_edge_accum(const int* __restrict__ ei,
                    const float* __restrict__ xl,
                    const float* __restrict__ e_in,
                    const float* __restrict__ emax,
                    float* __restrict__ denom, float* __restrict__ acc)
{
    const int edge = (blockIdx.x * 256 + threadIdx.x) >> 5;
    const int lane = threadIdx.x & 31;

    int s, d;
    if (edge < E_EDGES) { s = ei[edge]; d = ei[E_EDGES + edge]; }
    else                { s = edge - E_EDGES; d = s; }

    const float ex = __expf(e_in[edge] - emax[d]);
    if (lane == 0) atomicAdd(&denom[d], ex);

    const float4 xs = *(const float4*)&xl[s * C_DIM + lane * 4];
    float* o = &acc[d * C_DIM + lane * 4];
    atomicAdd(o + 0, ex * xs.x);
    atomicAdd(o + 1, ex * xs.y);
    atomicAdd(o + 2, ex * xs.z);
    atomicAdd(o + 3, ex * xs.w);
}

// ---------------------------------------------------------------------------
// Kernel 5: finalize.  out = acc/denom + res  (+ReLU for layer 1).
//           May write in place over acc (elementwise).
// ---------------------------------------------------------------------------
__global__ __launch_bounds__(256)
void gat_finalize(const float* __restrict__ acc, const float* __restrict__ res,
                  const float* __restrict__ denom, float* __restrict__ out,
                  int apply_relu)
{
    const int i = blockIdx.x * 256 + threadIdx.x;   // == N*128 exactly
    const int node = i >> 7;
    float v = acc[i] / denom[node] + res[i];
    if (apply_relu) v = v > 0.0f ? v : 0.0f;
    out[i] = v;
}

// ---------------------------------------------------------------------------
// Host side
// ---------------------------------------------------------------------------
extern "C" void kernel_launch(void* const* d_in, const int* in_sizes, int n_in,
                              void* d_out, int out_size, void* d_ws, size_t ws_size,
                              hipStream_t stream)
{
    const int*   ei  = (const int*)  d_in[0];   // [2, E]
    const float* emb = (const float*)d_in[1];   // [N, C]

    // layer params: Wl, bl, Wr, br, att, Wres, bias  (x2 layers)
    const float* l1[7]; const float* l2[7];
    for (int k = 0; k < 7; ++k) { l1[k] = (const float*)d_in[2 + k];
                                  l2[k] = (const float*)d_in[9 + k]; }

    // workspace layout (floats)
    const size_t NC = (size_t)N_NODES * C_DIM;     // 12.8M
    float* ws    = (float*)d_ws;
    float* xl    = ws;
    float* xr    = xl  + NC;
    float* res   = xr  + NC;
    float* acc   = res + NC;                       // doubles as h after layer 1
    float* e_b   = acc + NC;
    float* emax  = e_b + E_TOT;
    float* denom = emax + N_NODES;
    unsigned* wp = (unsigned*)(denom + N_NODES);   // packed weights (reused)
    unsigned* pl = wp;
    unsigned* pr = wp + WPACK_DW;
    unsigned* ps = wp + 2 * WPACK_DW;

    const dim3 blk(256);
    const dim3 gPack(WPACK_DW / 256);              // 32
    const dim3 gTransform(N_NODES / 16);           // 6250
    const dim3 gElem((N_NODES * C_DIM) / 256);     // 50000
    const dim3 gEdge(E_TOT / 8);                   // 92500 (8 waves/block)

    // ---- layer 1 ----
    gat_pack_weights<<<gPack, blk, 0, stream>>>(l1[0], l1[2], l1[5], pl, pr, ps);
    gat_transform<<<gTransform, blk, 0, stream>>>(emb, pl, pr, ps,
        l1[1], l1[3], l1[6], xl, xr, res);
    gat_init<<<gElem, blk, 0, stream>>>(emax, denom, acc);
    gat_edge_logits<<<gEdge, blk, 0, stream>>>(ei, xl, xr, l1[4], e_b, emax);
    gat_edge_accum<<<gEdge, blk, 0, stream>>>(ei, xl, e_b, emax, denom, acc);
    gat_finalize<<<gElem, blk, 0, stream>>>(acc, res, denom, acc, /*relu=*/1);

    // ---- layer 2 (h lives in acc) ----
    gat_pack_weights<<<gPack, blk, 0, stream>>>(l2[0], l2[2], l2[5], pl, pr, ps);
    gat_transform<<<gTransform, blk, 0, stream>>>(acc, pl, pr, ps,
        l2[1], l2[3], l2[6], xl, xr, res);
    gat_init<<<gElem, blk, 0, stream>>>(emax, denom, acc);
    gat_edge_logits<<<gEdge, blk, 0, stream>>>(ei, xl, xr, l2[4], e_b, emax);
    gat_edge_accum<<<gEdge, blk, 0, stream>>>(ei, xl, e_b, emax, denom, acc);
    gat_finalize<<<gElem, blk, 0, stream>>>(acc, res, denom, (float*)d_out,
                                            /*relu=*/0);
}
